// MCA_35356170780802
// MI455X (gfx1250) — compile-verified, matrix operand order checked by reading, not run on hardware
//
#include <hip/hip_runtime.h>
#include <hip/hip_bf16.h>

typedef __attribute__((ext_vector_type(16))) __bf16 v16bf;
typedef __attribute__((ext_vector_type(8)))  float  v8f;

#define BB 8
#define CC 64
#define NN 25600           // H*W
#define NBLK 1600          // NN/16
#define NCHUNK 64          // pixel chunks per (b,tile) in K1
#define K1_BLKS (NBLK / NCHUNK)   // 25
#define NB2 4              // 16-pixel blocks per wave in K2a/K2b
#define UNITS_PER_B (NBLK / NB2)  // 400

__device__ __forceinline__ v8f wmma_bf16(v16bf a, v16bf b, v8f c) {
  // D = A(16x32 bf16) * B(32x16 bf16) + C(16x16 f32)
  return __builtin_amdgcn_wmma_f32_16x16x32_bf16(false, a, false, b, (short)0, c,
                                                 false, false);
}

// fast softplus: 2 TRANS (v_exp_f32, v_log_f32) + 3 VALU
__device__ __forceinline__ float softplusf(float x) {
  return fmaxf(x, 0.0f) + __logf(1.0f + __expf(-fabsf(x)));
}

// A fragment: W [rows x ncols] row-major f32 -> 16x32 bf16 tile (tile-row trow, k-step ks)
// ISA layout: lane L holds row M=L&15; half=(L>>4): slots s<8 -> K=8*half+s, s>=8 -> K=16+8*half+(s-8)
__device__ __forceinline__ v16bf load_a_frag(const float* __restrict__ W, int ncols,
                                             int trow, int ks, int lane) {
  const int m  = trow * 16 + (lane & 15);
  const int kb = (lane >> 4) * 8;
  const float* row = W + (long)m * ncols + ks * 32;
  v16bf a;
#pragma unroll
  for (int s = 0; s < 16; ++s) {
    int col = (s < 8) ? (kb + s) : (16 + kb + (s - 8));
    a[s] = (__bf16)row[col];
  }
  return a;
}

// B fragment from activation X [C x NN] f32 (row-major over NN); mirrors A layout with N=lane&15
__device__ __forceinline__ v16bf load_b_frag_x(const float* __restrict__ X, long pix,
                                               int ks, int lane) {
  const int kb = (lane >> 4) * 8;
  const float* base = X + (long)ks * 32 * NN + pix;
  v16bf b;
#pragma unroll
  for (int s = 0; s < 16; ++s) {
    int ch = (s < 8) ? (kb + s) : (16 + kb + (s - 8));
    b[s] = (__bf16)base[(long)ch * NN];
  }
  return b;
}

__global__ void k0_zero(float* p, int n) {
  int i = blockIdx.x * blockDim.x + threadIdx.x;
  if (i < n) p[i] = 0.0f;
}

// K1: k = softplus(Wk*low+bk), v = Wv*low+bv via WMMA; accumulate per-head
// M[b][head-block]^T (M[c][m] = sum_n k[m,n] v[c,n]) and ksum via shfl reduce + atomics.
__global__ void __launch_bounds__(256, 1)
k1_kv(const float* __restrict__ low,
      const float* __restrict__ wk, const float* __restrict__ bk,
      const float* __restrict__ wv, const float* __restrict__ bv,
      float* __restrict__ Mmat, float* __restrict__ ksum) {
  const int lane = threadIdx.x & 31;
  const int wave = (blockIdx.x * blockDim.x + threadIdx.x) >> 5;
  const int chunk = wave % NCHUNK;
  const int bt = wave / NCHUNK;
  const int t = bt & 3;        // 16-channel output tile
  const int b = bt >> 2;
  const float* X = low + (long)b * CC * NN;

  v16bf ak0 = load_a_frag(wk, CC, t, 0, lane);
  v16bf ak1 = load_a_frag(wk, CC, t, 1, lane);
  v16bf av0 = load_a_frag(wv, CC, t, 0, lane);
  v16bf av1 = load_a_frag(wv, CC, t, 1, lane);

  const int half = lane >> 4;
  const int chbase = t * 16 + half * 8;   // this lane's head channel base; head = 2t+half
  float kb8[8], vb8[8];
#pragma unroll
  for (int r = 0; r < 8; ++r) { kb8[r] = bk[chbase + r]; vb8[r] = bv[chbase + r]; }

  float acc[64], ksacc[8];
#pragma unroll
  for (int i = 0; i < 64; ++i) acc[i] = 0.0f;
#pragma unroll
  for (int r = 0; r < 8; ++r) ksacc[r] = 0.0f;

  const int blk0 = chunk * K1_BLKS;
  for (int blk = blk0; blk < blk0 + K1_BLKS; ++blk) {
    const long pix = (long)blk * 16 + (lane & 15);
    v8f kc, vc;
#pragma unroll
    for (int r = 0; r < 8; ++r) { kc[r] = kb8[r]; vc[r] = vb8[r]; }
    {
      v16bf xb = load_b_frag_x(X, pix, 0, lane);
      kc = wmma_bf16(ak0, xb, kc);
      vc = wmma_bf16(av0, xb, vc);
    }
    {
      v16bf xb = load_b_frag_x(X, pix, 1, lane);
      kc = wmma_bf16(ak1, xb, kc);
      vc = wmma_bf16(av1, xb, vc);
    }
    float kk[8];
#pragma unroll
    for (int r = 0; r < 8; ++r) { kk[r] = softplusf(kc[r]); ksacc[r] += kk[r]; }
#pragma unroll
    for (int m = 0; m < 8; ++m)
#pragma unroll
      for (int c = 0; c < 8; ++c)
        acc[m * 8 + c] = fmaf(kk[m], vc[c], acc[m * 8 + c]);
  }

  // reduce over the 16 lanes of each half (xor masks < 16 stay within the half)
#pragma unroll
  for (int i = 0; i < 64; ++i) {
    float v = acc[i];
    v += __shfl_xor(v, 1, 32);
    v += __shfl_xor(v, 2, 32);
    v += __shfl_xor(v, 4, 32);
    v += __shfl_xor(v, 8, 32);
    acc[i] = v;
  }
#pragma unroll
  for (int r = 0; r < 8; ++r) {
    float v = ksacc[r];
    v += __shfl_xor(v, 1, 32);
    v += __shfl_xor(v, 2, 32);
    v += __shfl_xor(v, 4, 32);
    v += __shfl_xor(v, 8, 32);
    ksacc[r] = v;
  }
  if ((lane & 15) == 0) {
    const int head = t * 2 + half;
    float* Mb = Mmat + (long)b * 64 * 64;
    const int cb = head * 8;
#pragma unroll
    for (int m = 0; m < 8; ++m)
#pragma unroll
      for (int c = 0; c < 8; ++c)
        atomicAdd(&Mb[(cb + c) * 64 + (cb + m)], acc[m * 8 + c]);  // M[c][m] = attn[m][c]
#pragma unroll
    for (int r = 0; r < 8; ++r) atomicAdd(&ksum[b * 64 + cb + r], ksacc[r]);
  }
}

// K1.5: W2[b] = Wo @ M[b]; ksum_eps = ksum + 1e-6
__global__ void k15_prep(const float* __restrict__ wo, const float* __restrict__ Mmat,
                         const float* __restrict__ ksum, float* __restrict__ W2,
                         float* __restrict__ ksum_eps) {
  const int stride = blockDim.x * gridDim.x;
  for (int idx = blockIdx.x * blockDim.x + threadIdx.x; idx < BB * 64 * 64; idx += stride) {
    const int q = idx & 63;
    const int o = (idx >> 6) & 63;
    const int b = idx >> 12;
    const float* Mb = Mmat + (long)b * 4096;
    float s = 0.0f;
#pragma unroll 8
    for (int c = 0; c < 64; ++c) s = fmaf(wo[o * 64 + c], Mb[c * 64 + q], s);
    W2[idx] = s;
  }
  for (int idx = blockIdx.x * blockDim.x + threadIdx.x; idx < BB * 64; idx += stride)
    ksum_eps[idx] = ksum[idx] + 1e-6f;
}

// K2a: wave owns one branch (0: high -> q-tiles 0,1 / frag ks=0; 1: low -> q-tiles 2,3 / ks=1).
// q = softplus(Wq*x+bq); fold norm = 1/(q.(ksum+eps)) per head into q; pack bf16 fragment.
__global__ void __launch_bounds__(256, 1)
k2a_q(const float* __restrict__ high, const float* __restrict__ low,
      const float* __restrict__ wqh, const float* __restrict__ bqh,
      const float* __restrict__ wql, const float* __restrict__ bql,
      const float* __restrict__ ksum_eps, unsigned int* __restrict__ qbuf) {
  const int lane = threadIdx.x & 31;
  const int wave = (blockIdx.x * blockDim.x + threadIdx.x) >> 5;
  const int b = wave / (2 * UNITS_PER_B);
  const int rem = wave % (2 * UNITS_PER_B);
  const int branch = rem / UNITS_PER_B;          // 0 = high, 1 = low
  const int blkbase = (rem % UNITS_PER_B) * NB2;
  const int half = lane >> 4;

  const float* X  = (branch ? low : high) + (long)b * CC * NN;
  const float* wq = branch ? wql : wqh;
  const float* bq = branch ? bql : bqh;

  v16bf aq[2][2];
#pragma unroll
  for (int tt = 0; tt < 2; ++tt)
#pragma unroll
    for (int ks = 0; ks < 2; ++ks)
      aq[tt][ks] = load_a_frag(wq, CC, tt, ks, lane);

  float qbias[2][8], ksv[2][8];
#pragma unroll
  for (int tt = 0; tt < 2; ++tt) {
    const int head = 2 * (2 * branch + tt) + half;   // global head of this lane's rows
#pragma unroll
    for (int r = 0; r < 8; ++r) {
      qbias[tt][r] = bq[tt * 16 + half * 8 + r];
      ksv[tt][r]   = ksum_eps[b * 64 + head * 8 + r];
    }
  }

  for (int blk = blkbase; blk < blkbase + NB2; ++blk) {
    const long pix = (long)blk * 16 + (lane & 15);
    v16bf x0 = load_b_frag_x(X, pix, 0, lane);
    v16bf x1 = load_b_frag_x(X, pix, 1, lane);

    float qv[2][8];
#pragma unroll
    for (int tt = 0; tt < 2; ++tt) {
      v8f c;
#pragma unroll
      for (int r = 0; r < 8; ++r) c[r] = qbias[tt][r];
      c = wmma_bf16(aq[tt][0], x0, c);
      c = wmma_bf16(aq[tt][1], x1, c);
      float s = 0.0f;
#pragma unroll
      for (int r = 0; r < 8; ++r) { qv[tt][r] = softplusf(c[r]); s = fmaf(qv[tt][r], ksv[tt][r], s); }
      const float inv = 1.0f / s;
#pragma unroll
      for (int r = 0; r < 8; ++r) qv[tt][r] *= inv;
    }

    // pack the single bf16 B-fragment this wave owns (ks == branch) and store
    union { v16bf v; uint4 u[2]; } pk;
#pragma unroll
    for (int s = 0; s < 16; ++s)
      pk.v[s] = (s < 8) ? (__bf16)qv[0][s] : (__bf16)qv[1][s - 8];
    uint4* dst = reinterpret_cast<uint4*>(
        qbuf + ((((long)(b * NBLK + blk)) * 2 + branch) * 32 + lane) * 8);
    dst[0] = pk.u[0];
    dst[1] = pk.u[1];
  }
}

// K2b: out = W2[b] @ q' + bo; wave owns 2 of the 4 output tiles (thalf).
__global__ void __launch_bounds__(256, 1)
k2b_out(const unsigned int* __restrict__ qbuf, const float* __restrict__ W2,
        const float* __restrict__ bo, float* __restrict__ out) {
  const int lane = threadIdx.x & 31;
  const int wave = (blockIdx.x * blockDim.x + threadIdx.x) >> 5;
  const int b = wave / (2 * UNITS_PER_B);
  const int rem = wave % (2 * UNITS_PER_B);
  const int thalf = rem / UNITS_PER_B;            // output tiles {2*thalf, 2*thalf+1}
  const int blkbase = (rem % UNITS_PER_B) * NB2;
  const int half = lane >> 4;
  const float* W2b = W2 + (long)b * 4096;

  v16bf aw[2][2];
#pragma unroll
  for (int tt = 0; tt < 2; ++tt)
#pragma unroll
    for (int ks = 0; ks < 2; ++ks)
      aw[tt][ks] = load_a_frag(W2b, CC, 2 * thalf + tt, ks, lane);

  float obias[2][8];
#pragma unroll
  for (int tt = 0; tt < 2; ++tt)
#pragma unroll
    for (int r = 0; r < 8; ++r)
      obias[tt][r] = bo[(2 * thalf + tt) * 16 + half * 8 + r];

  for (int blk = blkbase; blk < blkbase + NB2; ++blk) {
    v16bf bf[2];
#pragma unroll
    for (int ks = 0; ks < 2; ++ks) {
      union { v16bf v; uint4 u[2]; } pk;
      const uint4* src = reinterpret_cast<const uint4*>(
          qbuf + ((((long)(b * NBLK + blk)) * 2 + ks) * 32 + lane) * 8);
      pk.u[0] = src[0];
      pk.u[1] = src[1];
      bf[ks] = pk.v;
    }
    float* ob = out + (long)b * CC * NN + (long)blk * 16 + (lane & 15);
#pragma unroll
    for (int tt = 0; tt < 2; ++tt) {
      v8f c;
#pragma unroll
      for (int r = 0; r < 8; ++r) c[r] = obias[tt][r];
      c = wmma_bf16(aw[tt][0], bf[0], c);
      c = wmma_bf16(aw[tt][1], bf[1], c);
      const int chb = (2 * thalf + tt) * 16 + half * 8;
#pragma unroll
      for (int r = 0; r < 8; ++r) ob[(long)(chb + r) * NN] = c[r];
    }
  }
}

extern "C" void kernel_launch(void* const* d_in, const int* in_sizes, int n_in,
                              void* d_out, int out_size, void* d_ws, size_t ws_size,
                              hipStream_t stream) {
  const float* high = (const float*)d_in[0];
  const float* low  = (const float*)d_in[1];
  const float* wqh  = (const float*)d_in[2];
  const float* bqh  = (const float*)d_in[3];
  const float* wql  = (const float*)d_in[4];
  const float* bql  = (const float*)d_in[5];
  const float* wk   = (const float*)d_in[6];
  const float* bk   = (const float*)d_in[7];
  const float* wv   = (const float*)d_in[8];
  const float* bv   = (const float*)d_in[9];
  const float* wo   = (const float*)d_in[10];
  const float* bo   = (const float*)d_in[11];
  float* out = (float*)d_out;

  // ws layout (f32): M[B*64*64] | ksum[B*64] | ksum_eps[B*64] | W2[B*64*64] | qbuf (bf16 frags)
  float* Mmat = (float*)d_ws;
  float* ksum = Mmat + BB * 64 * 64;
  float* ksum_eps = ksum + BB * 64;
  float* W2 = ksum_eps + BB * 64;
  unsigned int* qbuf = (unsigned int*)(W2 + BB * 64 * 64);  // B*NBLK*2*32*8 dwords ~ 26 MB

  const int nzero = BB * 64 * 64 + BB * 64;  // M and ksum are contiguous
  k0_zero<<<(nzero + 255) / 256, 256, 0, stream>>>(Mmat, nzero);
  k1_kv<<<(BB * 4 * NCHUNK) / 8, 256, 0, stream>>>(low, wk, bk, wv, bv, Mmat, ksum);
  k15_prep<<<64, 256, 0, stream>>>(wo, Mmat, ksum, W2, ksum_eps);
  k2a_q<<<(BB * 2 * UNITS_PER_B) / 8, 256, 0, stream>>>(high, low, wqh, bqh, wql, bql,
                                                        ksum_eps, qbuf);
  k2b_out<<<(BB * 2 * UNITS_PER_B) / 8, 256, 0, stream>>>(qbuf, W2, bo, out);
}
